// MultiAttentionLayer_86895778333353
// MI455X (gfx1250) — matrix operand order repaired and checked by reading, not compiled, not run on hardware
//
#include <hip/hip_runtime.h>

// MI455X / gfx1250, wave32. All matmuls via v_wmma_f32_16x16x32_bf16.

typedef __bf16 bf16_t;
typedef __attribute__((ext_vector_type(8)))  bf16_t v8bf;
typedef __attribute__((ext_vector_type(16))) bf16_t v16bf;
typedef __attribute__((ext_vector_type(8)))  float  v8f;

constexpr int Bc = 2, Sc = 2048, Ec = 1024, Hc = 16, Dc = 64;

__device__ __forceinline__ v8f wmma_bf16(v16bf a, v16bf b, v8f c) {
  return __builtin_amdgcn_wmma_f32_16x16x32_bf16(false, a, false, b, (short)0, c,
                                                 false, false);
}
__device__ __forceinline__ v8f zero8() {
  v8f z;
  #pragma unroll
  for (int t = 0; t < 8; ++t) z[t] = 0.0f;
  return z;
}
// A fragment (16x32, MxK): lane<16 holds K {0..7,16..23}, lane>=16 holds {8..15,24..31}
__device__ __forceinline__ v16bf frag_a(const bf16_t* rowPtr, int lane, int kcol) {
  const int off = kcol + ((lane < 16) ? 0 : 8);
  v8bf lo = *(const v8bf*)(rowPtr + off);
  v8bf hi = *(const v8bf*)(rowPtr + off + 16);
  return __builtin_shufflevector(lo, hi, 0,1,2,3,4,5,6,7,8,9,10,11,12,13,14,15);
}
// B fragment (32x16, KxN): lane<16 holds K {0..7,16..23} of column n, lane>=16 holds {8..15,24..31}
__device__ __forceinline__ v16bf frag_b(const bf16_t* colPtr, int lane, int kcol) {
  const int off = kcol + ((lane < 16) ? 0 : 8);
  v8bf lo = *(const v8bf*)(colPtr + off);
  v8bf hi = *(const v8bf*)(colPtr + off + 16);
  return __builtin_shufflevector(lo, hi, 0,1,2,3,4,5,6,7,8,9,10,11,12,13,14,15);
}

// ---------------- QKV projection: Out[b,h,s,d] = bf16( X[m,:]·W[n,:] + bias[n] )
__global__ __launch_bounds__(256) void mha_proj_gemm(
    const float* __restrict__ X, const float* __restrict__ W,
    const float* __restrict__ bias, bf16_t* __restrict__ Out) {
  constexpr int LD = 40;
  __shared__ __align__(16) bf16_t sA[128 * LD];
  __shared__ __align__(16) bf16_t sB[128 * LD];
  const int tid = threadIdx.x, lane = tid & 31, wv = tid >> 5;
  const int wm = wv & 1, wn = wv >> 1;
  const int m0 = blockIdx.y * 128, n0 = blockIdx.x * 128;

  v8f acc[4][2];
  #pragma unroll
  for (int mi = 0; mi < 4; ++mi)
    #pragma unroll
    for (int ni = 0; ni < 2; ++ni) acc[mi][ni] = zero8();

  for (int kk = 0; kk < Ec; kk += 32) {
    #pragma unroll
    for (int p = 0; p < 4; ++p) {
      const int id = tid + p * 256;
      const int r = id >> 3, c4 = (id & 7) * 4;
      float4 a = *(const float4*)(X + (size_t)(m0 + r) * Ec + kk + c4);
      float4 b = *(const float4*)(W + (size_t)(n0 + r) * Ec + kk + c4);
      bf16_t* da = sA + r * LD + c4;
      bf16_t* db = sB + r * LD + c4;
      da[0]=(bf16_t)a.x; da[1]=(bf16_t)a.y; da[2]=(bf16_t)a.z; da[3]=(bf16_t)a.w;
      db[0]=(bf16_t)b.x; db[1]=(bf16_t)b.y; db[2]=(bf16_t)b.z; db[3]=(bf16_t)b.w;
    }
    __syncthreads();
    v16bf bf[2];
    #pragma unroll
    for (int ni = 0; ni < 2; ++ni)
      bf[ni] = frag_b(sB + (wn * 32 + ni * 16 + (lane & 15)) * LD, lane, 0);
    #pragma unroll
    for (int mi = 0; mi < 4; ++mi) {
      v16bf af = frag_a(sA + (wm * 64 + mi * 16 + (lane & 15)) * LD, lane, 0);
      #pragma unroll
      for (int ni = 0; ni < 2; ++ni) acc[mi][ni] = wmma_bf16(af, bf[ni], acc[mi][ni]);
    }
    __syncthreads();
  }
  const int nl = lane & 15, rb = (lane < 16) ? 0 : 8;
  #pragma unroll
  for (int mi = 0; mi < 4; ++mi)
    #pragma unroll
    for (int ni = 0; ni < 2; ++ni) {
      const int n = n0 + wn * 32 + ni * 16 + nl;
      const float bs = bias[n];
      const int h = n >> 6, d = n & 63;
      #pragma unroll
      for (int i = 0; i < 8; ++i) {
        const int m = m0 + wm * 64 + mi * 16 + rb + i;
        const int b = m >> 11, s = m & (Sc - 1);
        Out[(((size_t)b * Hc + h) * Sc + s) * Dc + d] = (bf16_t)(acc[mi][ni][i] + bs);
      }
    }
}

// ---------------- scores: attn_raw[bh,i,j] = (Q·K^T)/8 ; skip fully-masked tiles
__global__ __launch_bounds__(256) void mha_scores(
    const bf16_t* __restrict__ Qp, const bf16_t* __restrict__ Kp,
    float* __restrict__ attn, const int* __restrict__ maskedPtr) {
  const int bh = blockIdx.z;
  const int i0 = blockIdx.y * 128, j0 = blockIdx.x * 128;
  if (*maskedPtr && (j0 > i0 + 127)) return;  // uniform across block
  constexpr int LD = 72;
  __shared__ __align__(16) bf16_t sQ[128 * LD];
  __shared__ __align__(16) bf16_t sK[128 * LD];
  const int tid = threadIdx.x, lane = tid & 31, wv = tid >> 5;
  const int wm = wv & 1, wn = wv >> 1;
  #pragma unroll
  for (int p = 0; p < 4; ++p) {
    const int id = tid + p * 256;
    const int r = id >> 3, c8 = (id & 7) * 8;
    *(v8bf*)(sQ + r * LD + c8) = *(const v8bf*)(Qp + ((size_t)bh * Sc + i0 + r) * Dc + c8);
    *(v8bf*)(sK + r * LD + c8) = *(const v8bf*)(Kp + ((size_t)bh * Sc + j0 + r) * Dc + c8);
  }
  __syncthreads();
  v8f acc[4][2];
  #pragma unroll
  for (int mi = 0; mi < 4; ++mi)
    #pragma unroll
    for (int ni = 0; ni < 2; ++ni) acc[mi][ni] = zero8();
  #pragma unroll
  for (int kk = 0; kk < 64; kk += 32) {
    v16bf bf[2];
    #pragma unroll
    for (int ni = 0; ni < 2; ++ni)
      bf[ni] = frag_b(sK + (wn * 32 + ni * 16 + (lane & 15)) * LD, lane, kk);
    #pragma unroll
    for (int mi = 0; mi < 4; ++mi) {
      v16bf af = frag_a(sQ + (wm * 64 + mi * 16 + (lane & 15)) * LD, lane, kk);
      #pragma unroll
      for (int ni = 0; ni < 2; ++ni) acc[mi][ni] = wmma_bf16(af, bf[ni], acc[mi][ni]);
    }
  }
  float* out = attn + (size_t)bh * Sc * Sc;
  const int nl = lane & 15, rb = (lane < 16) ? 0 : 8;
  #pragma unroll
  for (int mi = 0; mi < 4; ++mi)
    #pragma unroll
    for (int ni = 0; ni < 2; ++ni) {
      const int j = j0 + wn * 32 + ni * 16 + nl;
      #pragma unroll
      for (int i = 0; i < 8; ++i) {
        const int ii = i0 + wm * 64 + mi * 16 + rb + i;
        out[(size_t)ii * Sc + j] = acc[mi][ni][i] * 0.125f;
      }
    }
}

// ---------------- rowwise causal softmax (writes exact 0 for masked j)
__global__ __launch_bounds__(256) void mha_softmax(
    float* __restrict__ attn, const int* __restrict__ maskedPtr) {
  const int row = blockIdx.x;           // b*H*S + i flattened
  const int i = row & (Sc - 1);
  const int len = (*maskedPtr) ? (i + 1) : Sc;
  float* p = attn + (size_t)row * Sc;
  const int tid = threadIdx.x;
  __shared__ float red[256];
  float v[8], mx = -3.4e38f;
  #pragma unroll
  for (int t = 0; t < 8; ++t) {
    const int j = tid + t * 256;
    v[t] = (j < len) ? p[j] : -3.4e38f;
    mx = fmaxf(mx, v[t]);
  }
  red[tid] = mx; __syncthreads();
  for (int s = 128; s > 0; s >>= 1) {
    if (tid < s) red[tid] = fmaxf(red[tid], red[tid + s]);
    __syncthreads();
  }
  mx = red[0]; __syncthreads();
  float sum = 0.f;
  #pragma unroll
  for (int t = 0; t < 8; ++t) {
    const int j = tid + t * 256;
    v[t] = (j < len) ? expf(v[t] - mx) : 0.0f;
    sum += v[t];
  }
  red[tid] = sum; __syncthreads();
  for (int s = 128; s > 0; s >>= 1) {
    if (tid < s) red[tid] += red[tid + s];
    __syncthreads();
  }
  const float inv = 1.0f / red[0];
  #pragma unroll
  for (int t = 0; t < 8; ++t) p[tid + t * 256] = v[t] * inv;
}

// ---------------- attn · V  -> Ctx[b,s,h,d] (bf16); causal K-loop cap
__global__ __launch_bounds__(256) void mha_av_gemm(
    const float* __restrict__ attn, const bf16_t* __restrict__ Vp,
    bf16_t* __restrict__ Ctx, const int* __restrict__ maskedPtr) {
  const int bh = blockIdx.y;
  const int i0 = blockIdx.x * 128;
  const int kmax = (*maskedPtr) ? (i0 + 128) : Sc;
  constexpr int LD = 40;
  __shared__ __align__(16) bf16_t sA[128 * LD];
  __shared__ __align__(16) bf16_t sV[64 * LD];   // transposed: sV[d][j]
  const int tid = threadIdx.x, lane = tid & 31, wv = tid >> 5;
  const int wm = wv & 3, wn = wv >> 2;
  const float* arow = attn + (size_t)bh * Sc * Sc;
  v8f acc[2][2];
  #pragma unroll
  for (int mi = 0; mi < 2; ++mi)
    #pragma unroll
    for (int ni = 0; ni < 2; ++ni) acc[mi][ni] = zero8();

  for (int kk = 0; kk < kmax; kk += 32) {
    #pragma unroll
    for (int p = 0; p < 4; ++p) {
      const int id = tid + p * 256;
      const int r = id >> 3, c4 = (id & 7) * 4;
      float4 a = *(const float4*)(arow + (size_t)(i0 + r) * Sc + kk + c4);
      bf16_t* da = sA + r * LD + c4;
      da[0]=(bf16_t)a.x; da[1]=(bf16_t)a.y; da[2]=(bf16_t)a.z; da[3]=(bf16_t)a.w;
    }
    {
      const int j = tid >> 3, d0 = (tid & 7) * 8;
      v8bf x = *(const v8bf*)(Vp + ((size_t)bh * Sc + kk + j) * Dc + d0);
      #pragma unroll
      for (int u = 0; u < 8; ++u) sV[(d0 + u) * LD + j] = x[u];
    }
    __syncthreads();
    v16bf bf[2];
    #pragma unroll
    for (int ni = 0; ni < 2; ++ni)
      bf[ni] = frag_b(sV + (wn * 32 + ni * 16 + (lane & 15)) * LD, lane, 0);
    #pragma unroll
    for (int mi = 0; mi < 2; ++mi) {
      v16bf af = frag_a(sA + (wm * 32 + mi * 16 + (lane & 15)) * LD, lane, 0);
      #pragma unroll
      for (int ni = 0; ni < 2; ++ni) acc[mi][ni] = wmma_bf16(af, bf[ni], acc[mi][ni]);
    }
    __syncthreads();
  }
  const int b = bh >> 4, h = bh & 15;
  const int nl = lane & 15, rb = (lane < 16) ? 0 : 8;
  #pragma unroll
  for (int mi = 0; mi < 2; ++mi)
    #pragma unroll
    for (int ni = 0; ni < 2; ++ni) {
      const int d = wn * 32 + ni * 16 + nl;
      #pragma unroll
      for (int i = 0; i < 8; ++i) {
        const int s = i0 + wm * 32 + mi * 16 + rb + i;
        Ctx[(((size_t)b * Sc + s) * Hc + h) * Dc + d] = (bf16_t)acc[mi][ni][i];
      }
    }
}

// ---------------- output projection: Y[m,n] = Ctx[m,:]·Wo[n,:] + bo[n]  (f32 out)
__global__ __launch_bounds__(256) void mha_oproj_gemm(
    const bf16_t* __restrict__ Ctx, const float* __restrict__ Wo,
    const float* __restrict__ bo, float* __restrict__ Y) {
  constexpr int LD = 40;
  __shared__ __align__(16) bf16_t sA[128 * LD];
  __shared__ __align__(16) bf16_t sB[128 * LD];
  const int tid = threadIdx.x, lane = tid & 31, wv = tid >> 5;
  const int wm = wv & 1, wn = wv >> 1;
  const int m0 = blockIdx.y * 128, n0 = blockIdx.x * 128;
  v8f acc[4][2];
  #pragma unroll
  for (int mi = 0; mi < 4; ++mi)
    #pragma unroll
    for (int ni = 0; ni < 2; ++ni) acc[mi][ni] = zero8();

  for (int kk = 0; kk < Ec; kk += 32) {
    #pragma unroll
    for (int p = 0; p < 2; ++p) {
      const int id = tid + p * 256;
      const int r = id >> 2, c8 = (id & 3) * 8;
      *(v8bf*)(sA + r * LD + c8) = *(const v8bf*)(Ctx + (size_t)(m0 + r) * Ec + kk + c8);
    }
    #pragma unroll
    for (int p = 0; p < 4; ++p) {
      const int id = tid + p * 256;
      const int r = id >> 3, c4 = (id & 7) * 4;
      float4 b = *(const float4*)(Wo + (size_t)(n0 + r) * Ec + kk + c4);
      bf16_t* db = sB + r * LD + c4;
      db[0]=(bf16_t)b.x; db[1]=(bf16_t)b.y; db[2]=(bf16_t)b.z; db[3]=(bf16_t)b.w;
    }
    __syncthreads();
    v16bf bf[2];
    #pragma unroll
    for (int ni = 0; ni < 2; ++ni)
      bf[ni] = frag_b(sB + (wn * 32 + ni * 16 + (lane & 15)) * LD, lane, 0);
    #pragma unroll
    for (int mi = 0; mi < 4; ++mi) {
      v16bf af = frag_a(sA + (wm * 64 + mi * 16 + (lane & 15)) * LD, lane, 0);
      #pragma unroll
      for (int ni = 0; ni < 2; ++ni) acc[mi][ni] = wmma_bf16(af, bf[ni], acc[mi][ni]);
    }
    __syncthreads();
  }
  const int nl = lane & 15, rb = (lane < 16) ? 0 : 8;
  #pragma unroll
  for (int mi = 0; mi < 4; ++mi)
    #pragma unroll
    for (int ni = 0; ni < 2; ++ni) {
      const int n = n0 + wn * 32 + ni * 16 + nl;
      const float bs = bo[n];
      #pragma unroll
      for (int i = 0; i < 8; ++i) {
        const int m = m0 + wm * 64 + mi * 16 + rb + i;
        Y[(size_t)m * Ec + n] = acc[mi][ni][i] + bs;
      }
    }
}

// ---------------- residual + LayerNorm (in-place on Y rows)
__global__ __launch_bounds__(256) void mha_ln(
    float* __restrict__ Y, const float* __restrict__ Qin,
    const float* __restrict__ gamma, const float* __restrict__ beta) {
  const int row = blockIdx.x, tid = threadIdx.x;
  float* yr = Y + (size_t)row * Ec;
  const float* qr = Qin + (size_t)row * Ec;
  __shared__ float r1[256], r2[256];
  float x[4], s = 0.f, sq = 0.f;
  #pragma unroll
  for (int t = 0; t < 4; ++t) {
    const int e = tid + t * 256;
    x[t] = yr[e] + qr[e];
    s += x[t]; sq += x[t] * x[t];
  }
  r1[tid] = s; r2[tid] = sq; __syncthreads();
  for (int st = 128; st > 0; st >>= 1) {
    if (tid < st) { r1[tid] += r1[tid + st]; r2[tid] += r2[tid + st]; }
    __syncthreads();
  }
  const float mu = r1[0] * (1.0f / Ec);
  const float var = r2[0] * (1.0f / Ec) - mu * mu;
  const float rs = rsqrtf(var + 1e-5f);
  #pragma unroll
  for (int t = 0; t < 4; ++t) {
    const int e = tid + t * 256;
    yr[e] = (x[t] - mu) * rs * gamma[e] + beta[e];
  }
}

extern "C" void kernel_launch(void* const* d_in, const int* in_sizes, int n_in,
                              void* d_out, int out_size, void* d_ws, size_t ws_size,
                              hipStream_t stream) {
  const float* q    = (const float*)d_in[0];
  const float* k    = (const float*)d_in[1];
  const float* v    = (const float*)d_in[2];
  const float* Wq   = (const float*)d_in[3];
  const float* bq   = (const float*)d_in[4];
  const float* Wk   = (const float*)d_in[5];
  const float* bk   = (const float*)d_in[6];
  const float* Wv   = (const float*)d_in[7];
  const float* bv   = (const float*)d_in[8];
  const float* Wo   = (const float*)d_in[9];
  const float* bo   = (const float*)d_in[10];
  const float* gam  = (const float*)d_in[11];
  const float* bet  = (const float*)d_in[12];
  const int*   mskd = (const int*)d_in[13];

  float* y_out    = (float*)d_out;
  float* attn_out = y_out + (size_t)Bc * Sc * Ec;

  const size_t PER = (size_t)Bc * Hc * Sc * Dc;   // 4,194,304 elems
  bf16_t* Qp  = (bf16_t*)d_ws;
  bf16_t* Kp  = Qp + PER;
  bf16_t* Vp  = Kp + PER;
  bf16_t* Ctx = Vp + PER;                          // 32 MB total workspace

  dim3 blk(256);
  dim3 gP(Ec / 128, (Bc * Sc) / 128);              // 8 x 32
  mha_proj_gemm<<<gP, blk, 0, stream>>>(q, Wq, bq, Qp);
  mha_proj_gemm<<<gP, blk, 0, stream>>>(k, Wk, bk, Kp);
  mha_proj_gemm<<<gP, blk, 0, stream>>>(v, Wv, bv, Vp);

  dim3 gS(Sc / 128, Sc / 128, Bc * Hc);            // 16 x 16 x 32
  mha_scores<<<gS, blk, 0, stream>>>(Qp, Kp, attn_out, mskd);

  mha_softmax<<<dim3(Bc * Hc * Sc), blk, 0, stream>>>(attn_out, mskd);

  mha_av_gemm<<<dim3(Sc / 128, Bc * Hc), blk, 0, stream>>>(attn_out, Vp, Ctx, mskd);

  mha_oproj_gemm<<<dim3(Ec / 128, (Bc * Sc) / 128), blk, 0, stream>>>(Ctx, Wo, bo, y_out);

  mha_ln<<<dim3(Bc * Sc), blk, 0, stream>>>(y_out, q, gam, bet);
}
